// Qwen3NextSparseMoeBlock_25245817766043
// MI455X (gfx1250) — matrix operand order repaired and checked by reading, not compiled, not run on hardware
//
#include <hip/hip_runtime.h>

typedef __attribute__((ext_vector_type(16))) __bf16   v16bf;
typedef __attribute__((ext_vector_type(8)))  float    v8f;
typedef __attribute__((ext_vector_type(4)))  float    f32x4;
typedef __attribute__((ext_vector_type(4)))  unsigned u32x4;

union BF16Tile { u32x4 d[2]; unsigned u[8]; v16bf v; };

// Pack two floats into packed bf16 pair (round-to-nearest) in 3 VALU ops.
static __device__ __forceinline__ unsigned pk2(float lo, float hi) {
    unsigned u0 = __builtin_bit_cast(unsigned, lo) + 0x8000u;
    unsigned u1 = __builtin_bit_cast(unsigned, hi) + 0x8000u;
    return __builtin_amdgcn_perm(u1, u0, 0x07060302u);
}

static __device__ __forceinline__ __bf16 f2bf(float f) {
    unsigned u = __builtin_bit_cast(unsigned, f) + 0x8000u;
    unsigned short s = (unsigned short)(u >> 16);
    return __builtin_bit_cast(__bf16, s);
}

// ---- global->LDS async copy (CDNA5): one 16B chunk per lane, ASYNCcnt ----
static __device__ __forceinline__ void async_copy16(unsigned lds, const void* g) {
    asm volatile("global_load_async_to_lds_b128 %0, %1, off"
                 :: "v"(lds), "v"(g) : "memory");
}
#define WAIT_ASYNC(n) asm volatile("s_wait_asynccnt " #n ::: "memory")

// A fragment (16x32 bf16): p = rowbase + subK*64 + 32*half ; chunks +0,+16 B.
static __device__ __forceinline__ v16bf lds_frag_a(const char* p) {
    BF16Tile r;
    r.d[0] = *(const u32x4*)(p);
    r.d[1] = *(const u32x4*)(p + 16);
    return r.v;
}
// B fragment (32x16 bf16): p = rowbase + subK*64 + 32*half ; chunks +0,+16 B.
static __device__ __forceinline__ v16bf lds_frag_b(const char* p) {
    BF16Tile r;
    r.d[0] = *(const u32x4*)(p);
    r.d[1] = *(const u32x4*)(p + 16);
    return r.v;
}

// Register-direct loads (fp32 fallback path).
static __device__ __forceinline__ v16bf load_a16(const __bf16* p) {
    BF16Tile r;
    r.d[0] = *(const u32x4*)(p);
    r.d[1] = *(const u32x4*)(p + 8);
    return r.v;
}
static __device__ __forceinline__ v16bf load_b32(const float* p) {
    f32x4 b0 = *(const f32x4*)(p);
    f32x4 b1 = *(const f32x4*)(p + 4);
    f32x4 b2 = *(const f32x4*)(p + 8);
    f32x4 b3 = *(const f32x4*)(p + 12);
    BF16Tile r;
    r.u[0] = pk2(b0[0], b0[1]); r.u[1] = pk2(b0[2], b0[3]);
    r.u[2] = pk2(b1[0], b1[1]); r.u[3] = pk2(b1[2], b1[3]);
    r.u[4] = pk2(b2[0], b2[1]); r.u[5] = pk2(b2[2], b2[3]);
    r.u[6] = pk2(b3[0], b3[1]); r.u[7] = pk2(b3[2], b3[3]);
    return r.v;
}

#define WMMA_BF16(A, B, C) \
    __builtin_amdgcn_wmma_f32_16x16x32_bf16(false, (A), false, (B), (short)0, (C), false, false)

#define KSTEP   64
#define RSTRIDE 144  /* LDS row stride: 128B data + 16B pad (bank spread) */
#define A_BUF   (128 * RSTRIDE)
#define B_BUF   (64 * RSTRIDE)

// ---------------------------------------------------------------------------
// Elementwise fp32 -> bf16 (8 elems/thread), memory bound.
// ---------------------------------------------------------------------------
__global__ __launch_bounds__(256) void
cvt_bf16_kernel(const float* __restrict__ src, __bf16* __restrict__ dst, size_t n)
{
    size_t i = ((size_t)blockIdx.x * blockDim.x + threadIdx.x) * 8;
    if (i >= n) return;
    f32x4 a = *(const f32x4*)(src + i);
    f32x4 b = *(const f32x4*)(src + i + 4);
    u32x4 o;
    o[0] = pk2(a[0], a[1]); o[1] = pk2(a[2], a[3]);
    o[2] = pk2(b[0], b[1]); o[3] = pk2(b[2], b[3]);
    *(u32x4*)(dst + i) = o;
}

// ---------------------------------------------------------------------------
// Router: 16 expert logits + shared sigmoid gate; softmax/top-2/renorm into
// dense combine[T,E]; writes bf16 copy of x. One wave per token.
// ---------------------------------------------------------------------------
__global__ __launch_bounds__(128) void
router_kernel(const float* __restrict__ x, const float* __restrict__ gate_w,
              const float* __restrict__ sgate,
              float* __restrict__ combine, float* __restrict__ gout,
              __bf16* __restrict__ xbf, int T, int Hdim, int Eexp)
{
    const int lane = threadIdx.x & 31;
    const int wave = threadIdx.x >> 5;
    const int t = blockIdx.x * (blockDim.x >> 5) + wave;
    if (t >= T) return;

    const int chunk = Hdim >> 5;
    const int h0 = lane * chunk;
    const float* xr = x + (size_t)t * Hdim + h0;
    __bf16* xw = xbf + (size_t)t * Hdim + h0;

    float acc[17];
#pragma unroll
    for (int i = 0; i < 17; ++i) acc[i] = 0.f;

    for (int c = 0; c < chunk; c += 8) {
        f32x4 a = *(const f32x4*)(xr + c);
        f32x4 b = *(const f32x4*)(xr + c + 4);
        u32x4 o;
        o[0] = pk2(a[0], a[1]); o[1] = pk2(a[2], a[3]);
        o[2] = pk2(b[0], b[1]); o[3] = pk2(b[2], b[3]);
        *(u32x4*)(xw + c) = o;
#pragma unroll
        for (int e = 0; e < 16; ++e) {
            const float* gr = gate_w + (size_t)e * Hdim + h0 + c;
            acc[e] += a[0]*gr[0] + a[1]*gr[1] + a[2]*gr[2] + a[3]*gr[3]
                    + b[0]*gr[4] + b[1]*gr[5] + b[2]*gr[6] + b[3]*gr[7];
        }
        const float* sr = sgate + h0 + c;
        acc[16] += a[0]*sr[0] + a[1]*sr[1] + a[2]*sr[2] + a[3]*sr[3]
                 + b[0]*sr[4] + b[1]*sr[5] + b[2]*sr[6] + b[3]*sr[7];
    }
#pragma unroll
    for (int i = 0; i < 17; ++i) {
        float v = acc[i];
#pragma unroll
        for (int off = 16; off >= 1; off >>= 1) v += __shfl_xor(v, off, 32);
        acc[i] = v;
    }
    if (lane == 0) {
        float mx = acc[0];
#pragma unroll
        for (int e = 1; e < 16; ++e) mx = fmaxf(mx, acc[e]);
        float p[16];
#pragma unroll
        for (int e = 0; e < 16; ++e) p[e] = __expf(acc[e] - mx);
        int i1 = 0; float v1 = p[0];
#pragma unroll
        for (int e = 1; e < 16; ++e) if (p[e] > v1) { v1 = p[e]; i1 = e; }
        int i2 = 0; float v2 = -1.f;
#pragma unroll
        for (int e = 0; e < 16; ++e) if (e != i1 && p[e] > v2) { v2 = p[e]; i2 = e; }
        float inv = 1.f / (v1 + v2);
#pragma unroll
        for (int e = 0; e < 16; ++e) combine[(size_t)t * Eexp + e] = 0.f;
        combine[(size_t)t * Eexp + i1] = v1 * inv;
        combine[(size_t)t * Eexp + i2] = v2 * inv;
        gout[t] = 1.f / (1.f + __expf(-acc[16]));
    }
}

// ---------------------------------------------------------------------------
// Up projection, bf16 weights, triple-buffered async LDS staging, one barrier
// per stage. WG tile 128(tok) x 64(N); 16 WMMAs per K=64 stage.
// ---------------------------------------------------------------------------
__global__ __launch_bounds__(256) void
moe_up_lds(const __bf16* __restrict__ xbf,
           const __bf16* __restrict__ Wg, const __bf16* __restrict__ Wu,
           const float* __restrict__ combine, __bf16* __restrict__ midout,
           int T, int Hdim, int Nout, int Eexp, int mode)
{
    __shared__ __align__(16) char smem[3 * A_BUF + 6 * B_BUF];   // 108 KB
    const unsigned A_OFF  = 0;
    const unsigned BG_OFF = 3 * A_BUF;
    const unsigned BU_OFF = BG_OFF + 3 * B_BUF;
    const unsigned sbase  = (unsigned)(size_t)(void*)&smem[0];

    const int tid  = threadIdx.x;
    const int lane = tid & 31;
    const int wave = tid >> 5;
    const int half = lane >> 4;
    const int m    = lane & 15;

    const int tokWG = blockIdx.x * 128;
    const int tok0  = tokWG + wave * 16;
    const int n0    = blockIdx.y * 64;
    const int e     = (mode == 0) ? blockIdx.z : 0;

    const __bf16* wg = Wg + (size_t)e * Nout * Hdim;
    const __bf16* wu = Wu + (size_t)e * Nout * Hdim;
    __bf16* mout = midout + (size_t)e * T * Nout;

    auto issue = [&](int k0, int b) {
        // A: 128 rows x 8 chunks = 1024 -> 4/thread
#pragma unroll
        for (int i = 0; i < 4; ++i) {
            int c = tid + i * 256, r = c >> 3, q = c & 7;
            async_copy16(sbase + A_OFF + b * A_BUF + r * RSTRIDE + q * 16,
                         xbf + (size_t)(tokWG + r) * Hdim + k0 + q * 8);
        }
        // Bg/Bu: 64 rows x 8 chunks = 512 each -> 2/thread each
#pragma unroll
        for (int i = 0; i < 2; ++i) {
            int c = tid + i * 256, r = c >> 3, q = c & 7;
            size_t go = (size_t)(n0 + r) * Hdim + k0 + q * 8;
            async_copy16(sbase + BG_OFF + b * B_BUF + r * RSTRIDE + q * 16, wg + go);
            async_copy16(sbase + BU_OFF + b * B_BUF + r * RSTRIDE + q * 16, wu + go);
        }
    };

    v8f accg[4] = {};
    v8f accu[4] = {};

    const int nStages = Hdim / KSTEP;
    int bCur = 0, bNxt = 1;
    issue(0, 0);
    for (int s = 0; s < nStages; ++s) {
        if (s + 1 < nStages) {
            issue((s + 1) * KSTEP, bNxt);
            WAIT_ASYNC(8);            // leave only next stage's copies in flight
        } else {
            WAIT_ASYNC(0);
        }
        __syncthreads();

        const char* ab  = smem + A_OFF  + bCur * A_BUF
                        + (wave * 16 + m) * RSTRIDE + 32 * half;
        const char* bgb = smem + BG_OFF + bCur * B_BUF + m * RSTRIDE + 32 * half;
        const char* bub = smem + BU_OFF + bCur * B_BUF + m * RSTRIDE + 32 * half;
#pragma unroll
        for (int ks = 0; ks < 2; ++ks) {
            const int kb = ks * 64;                 // byte offset of K sub-step
            v16bf a = lds_frag_a(ab + kb);
            v16bf bg[4], bu[4];
#pragma unroll
            for (int nb = 0; nb < 4; ++nb) {
                bg[nb] = lds_frag_b(bgb + kb + nb * 16 * RSTRIDE);
                bu[nb] = lds_frag_b(bub + kb + nb * 16 * RSTRIDE);
            }
#pragma unroll
            for (int nb = 0; nb < 4; ++nb) {
                accg[nb] = WMMA_BF16(a, bg[nb], accg[nb]);
                accu[nb] = WMMA_BF16(a, bu[nb], accu[nb]);
            }
        }
        bCur = bNxt;
        bNxt = (bNxt == 2) ? 0 : bNxt + 1;
    }

    float sc[8];
#pragma unroll
    for (int r = 0; r < 8; ++r) {
        int t = tok0 + half * 8 + r;
        sc[r] = (mode == 0) ? combine[(size_t)t * Eexp + e] : 1.0f;
    }
#pragma unroll
    for (int nb = 0; nb < 4; ++nb) {
        int n = n0 + nb * 16 + m;
#pragma unroll
        for (int r = 0; r < 8; ++r) {
            int t = tok0 + half * 8 + r;
            float g = accg[nb][r];
            float u = accu[nb][r];
            float silu = g / (1.0f + __expf(-g));
            mout[(size_t)t * Nout + n] = f2bf(silu * u * sc[r]);
        }
    }
}

// ---------------------------------------------------------------------------
// Down projection, bf16, triple-buffered async LDS staging.
// mode 0: write f32 moe_out. mode 1 (shared): out = moe_in + g[t]*acc.
// ---------------------------------------------------------------------------
__global__ __launch_bounds__(256) void
moe_down_lds(const __bf16* __restrict__ Amid, const __bf16* __restrict__ W,
             const float* __restrict__ moe_in, const float* __restrict__ gvec,
             float* __restrict__ outp,
             int T, int Kdim, int Hdim, int nE, int mode)
{
    __shared__ __align__(16) char smem[3 * A_BUF + 3 * B_BUF];   // 81 KB
    const unsigned A_OFF = 0;
    const unsigned B_OFF = 3 * A_BUF;
    const unsigned sbase = (unsigned)(size_t)(void*)&smem[0];

    const int tid  = threadIdx.x;
    const int lane = tid & 31;
    const int wave = tid >> 5;
    const int half = lane >> 4;
    const int m    = lane & 15;

    const int tokWG = blockIdx.x * 128;
    const int tok0  = tokWG + wave * 16;
    const int n0    = blockIdx.y * 64;

    const int spe = Kdim / KSTEP;           // stages per expert
    const int nStages = nE * spe;

    auto issue = [&](int s, int b) {
        int e  = s / spe;
        int k0 = (s - e * spe) * KSTEP;
        const __bf16* ae = Amid + (size_t)e * T * Kdim;
        const __bf16* be = W    + (size_t)e * Hdim * Kdim;
#pragma unroll
        for (int i = 0; i < 4; ++i) {
            int c = tid + i * 256, r = c >> 3, q = c & 7;
            async_copy16(sbase + A_OFF + b * A_BUF + r * RSTRIDE + q * 16,
                         ae + (size_t)(tokWG + r) * Kdim + k0 + q * 8);
        }
#pragma unroll
        for (int i = 0; i < 2; ++i) {
            int c = tid + i * 256, r = c >> 3, q = c & 7;
            async_copy16(sbase + B_OFF + b * B_BUF + r * RSTRIDE + q * 16,
                         be + (size_t)(n0 + r) * Kdim + k0 + q * 8);
        }
    };

    v8f acc[4] = {};

    const int nS = nStages;
    int bCur = 0, bNxt = 1;
    issue(0, 0);
    for (int s = 0; s < nS; ++s) {
        if (s + 1 < nS) {
            issue(s + 1, bNxt);
            WAIT_ASYNC(6);
        } else {
            WAIT_ASYNC(0);
        }
        __syncthreads();

        const char* ap = smem + A_OFF + bCur * A_BUF
                       + (wave * 16 + m) * RSTRIDE + 32 * half;
        const char* bp = smem + B_OFF + bCur * B_BUF + m * RSTRIDE + 32 * half;
#pragma unroll
        for (int ks = 0; ks < 2; ++ks) {
            const int kb = ks * 64;
            v16bf a = lds_frag_a(ap + kb);
            v16bf bb[4];
#pragma unroll
            for (int nb = 0; nb < 4; ++nb)
                bb[nb] = lds_frag_b(bp + kb + nb * 16 * RSTRIDE);
#pragma unroll
            for (int nb = 0; nb < 4; ++nb)
                acc[nb] = WMMA_BF16(a, bb[nb], acc[nb]);
        }
        bCur = bNxt;
        bNxt = (bNxt == 2) ? 0 : bNxt + 1;
    }

#pragma unroll
    for (int nb = 0; nb < 4; ++nb) {
        int n = n0 + nb * 16 + m;
#pragma unroll
        for (int r = 0; r < 8; ++r) {
            int t = tok0 + half * 8 + r;
            size_t idx = (size_t)t * Hdim + n;
            if (mode == 0) outp[idx] = acc[nb][r];
            else           outp[idx] = moe_in[idx] + gvec[t] * acc[nb][r];
        }
    }
}

// ---------------------------------------------------------------------------
// Fallback GEMMs for fp32 weights streamed from HBM (in-register bf16 cvt).
// ---------------------------------------------------------------------------
__global__ __launch_bounds__(256) void
moe_up_direct(const __bf16* __restrict__ xbf,
              const float* __restrict__ Wg, const float* __restrict__ Wu,
              const float* __restrict__ combine, __bf16* __restrict__ midout,
              int T, int Hdim, int Nout, int Eexp, int mode)
{
    const int lane = threadIdx.x & 31;
    const int wave = threadIdx.x >> 5;
    const int half = lane >> 4;
    const int m    = lane & 15;

    const int tok0 = blockIdx.x * 128 + wave * 16;
    const int n0   = blockIdx.y * 64;
    const int e    = (mode == 0) ? blockIdx.z : 0;

    __bf16* mout = midout + (size_t)e * T * Nout;
    v8f accg[4] = {};
    v8f accu[4] = {};

    const __bf16* arow = xbf + (size_t)(tok0 + m) * Hdim + 16 * half;
    const size_t  boff = (size_t)e * Nout * Hdim + (size_t)(n0 + m) * Hdim + 16 * half;

    for (int k0 = 0; k0 < Hdim; k0 += 32) {
        v16bf a = load_a16(arow + k0);
#pragma unroll
        for (int nb = 0; nb < 4; ++nb) {
            size_t bo = boff + (size_t)(nb * 16) * Hdim + k0;
            v16bf bg = load_b32(Wg + bo);
            accg[nb] = WMMA_BF16(a, bg, accg[nb]);
            v16bf bu = load_b32(Wu + bo);
            accu[nb] = WMMA_BF16(a, bu, accu[nb]);
        }
    }
    float sc[8];
#pragma unroll
    for (int r = 0; r < 8; ++r) {
        int t = tok0 + half * 8 + r;
        sc[r] = (mode == 0) ? combine[(size_t)t * Eexp + e] : 1.0f;
    }
#pragma unroll
    for (int nb = 0; nb < 4; ++nb) {
        int n = n0 + nb * 16 + m;
#pragma unroll
        for (int r = 0; r < 8; ++r) {
            int t = tok0 + half * 8 + r;
            float g = accg[nb][r];
            float u = accu[nb][r];
            float silu = g / (1.0f + __expf(-g));
            mout[(size_t)t * Nout + n] = f2bf(silu * u * sc[r]);
        }
    }
}

__global__ __launch_bounds__(256) void
moe_down_direct(const __bf16* __restrict__ Amid, const float* __restrict__ W,
                const float* __restrict__ moe_in, const float* __restrict__ gvec,
                float* __restrict__ outp,
                int T, int Kdim, int Hdim, int nE, int mode)
{
    const int lane = threadIdx.x & 31;
    const int wave = threadIdx.x >> 5;
    const int half = lane >> 4;
    const int m    = lane & 15;

    const int tok0 = blockIdx.x * 128 + wave * 16;
    const int n0   = blockIdx.y * 64;

    v8f acc[4] = {};
    for (int e = 0; e < nE; ++e) {
        const __bf16* arow = Amid + (size_t)e * T * Kdim
                                  + (size_t)(tok0 + m) * Kdim + 16 * half;
        const size_t  boff = (size_t)e * Hdim * Kdim
                                  + (size_t)(n0 + m) * Kdim + 16 * half;
        for (int k0 = 0; k0 < Kdim; k0 += 32) {
            v16bf a = load_a16(arow + k0);
#pragma unroll
            for (int nb = 0; nb < 4; ++nb) {
                v16bf b = load_b32(W + boff + (size_t)(nb * 16) * Kdim + k0);
                acc[nb] = WMMA_BF16(a, b, acc[nb]);
            }
        }
    }
#pragma unroll
    for (int nb = 0; nb < 4; ++nb) {
        int n = n0 + nb * 16 + m;
#pragma unroll
        for (int r = 0; r < 8; ++r) {
            int t = tok0 + half * 8 + r;
            size_t idx = (size_t)t * Hdim + n;
            if (mode == 0) outp[idx] = acc[nb][r];
            else           outp[idx] = moe_in[idx] + gvec[t] * acc[nb][r];
        }
    }
}

// ---------------------------------------------------------------------------
extern "C" void kernel_launch(void* const* d_in, const int* in_sizes, int n_in,
                              void* d_out, int out_size, void* d_ws, size_t ws_size,
                              hipStream_t stream)
{
    (void)in_sizes; (void)n_in; (void)out_size;
    const int T = 2048, H = 2048, E = 16, I = 512, IS = 2048;

    const float* x   = (const float*)d_in[0];
    const float* gw  = (const float*)d_in[1];
    const float* w1  = (const float*)d_in[2];
    const float* w2  = (const float*)d_in[3];
    const float* w3  = (const float*)d_in[4];
    const float* wsg = (const float*)d_in[5];
    const float* wsu = (const float*)d_in[6];
    const float* wsd = (const float*)d_in[7];
    const float* sgw = (const float*)d_in[8];
    float* outp = (float*)d_out;

    char* ws = (char*)d_ws;
    size_t off = 0;
    auto alloc = [&](size_t bytes) -> void* {
        void* p = ws + off;
        off = (off + bytes + 255) & ~(size_t)255;
        return p;
    };
    __bf16* xbf     = (__bf16*)alloc((size_t)T * H * 2);
    float*  combine = (float*) alloc((size_t)T * E * 4);
    float*  gvec    = (float*) alloc((size_t)T * 4);
    __bf16* hmidE   = (__bf16*)alloc((size_t)E * T * I * 2);
    __bf16* smid    = (__bf16*)alloc((size_t)T * IS * 2);
    float*  moe     = (float*) alloc((size_t)T * H * 4);

    const size_t nw  = (size_t)E * I * H;
    const size_t nws = (size_t)IS * H;
    __bf16* w1b  = (__bf16*)alloc(nw  * 2);
    __bf16* w3b  = (__bf16*)alloc(nw  * 2);
    __bf16* w2b  = (__bf16*)alloc(nw  * 2);
    __bf16* wsgb = (__bf16*)alloc(nws * 2);
    __bf16* wsub = (__bf16*)alloc(nws * 2);
    __bf16* wsdb = (__bf16*)alloc(nws * 2);
    const bool useB16 = (off <= ws_size);

    router_kernel<<<T / 4, 128, 0, stream>>>(x, gw, sgw, combine, gvec, xbf, T, H, E);

    dim3 gUpE(T / 128, I / 64, E);
    dim3 gUpS(T / 128, IS / 64, 1);
    dim3 gD(T / 128, H / 64, 1);

    if (useB16) {
        int blkW  = (int)(nw  / 8 / 256);
        int blkWs = (int)(nws / 8 / 256);
        cvt_bf16_kernel<<<blkW,  256, 0, stream>>>(w1,  w1b,  nw);
        cvt_bf16_kernel<<<blkW,  256, 0, stream>>>(w3,  w3b,  nw);
        cvt_bf16_kernel<<<blkW,  256, 0, stream>>>(w2,  w2b,  nw);
        cvt_bf16_kernel<<<blkWs, 256, 0, stream>>>(wsg, wsgb, nws);
        cvt_bf16_kernel<<<blkWs, 256, 0, stream>>>(wsu, wsub, nws);
        cvt_bf16_kernel<<<blkWs, 256, 0, stream>>>(wsd, wsdb, nws);

        moe_up_lds<<<gUpE, 256, 0, stream>>>(xbf, w1b, w3b, combine, hmidE, T, H, I, E, 0);
        moe_up_lds<<<gUpS, 256, 0, stream>>>(xbf, wsgb, wsub, nullptr, smid, T, H, IS, E, 1);
        moe_down_lds<<<gD, 256, 0, stream>>>(hmidE, w2b, nullptr, nullptr, moe, T, I, H, E, 0);
        moe_down_lds<<<gD, 256, 0, stream>>>(smid, wsdb, moe, gvec, outp, T, IS, H, 1, 1);
    } else {
        moe_up_direct<<<gUpE, 256, 0, stream>>>(xbf, w1, w3, combine, hmidE, T, H, I, E, 0);
        moe_up_direct<<<gUpS, 256, 0, stream>>>(xbf, wsg, wsu, nullptr, smid, T, H, IS, E, 1);
        moe_down_direct<<<gD, 256, 0, stream>>>(hmidE, w2, nullptr, nullptr, moe, T, I, H, E, 0);
        moe_down_direct<<<gD, 256, 0, stream>>>(smid, wsd, moe, gvec, outp, T, IS, H, 1, 1);
    }
}
